// MAB_26542897889892
// MI455X (gfx1250) — compile-verified, matrix operand order checked by reading, not run on hardware
//
#include <hip/hip_runtime.h>
#include <hip/hip_bf16.h>
#include <math.h>

#define B_    32
#define LQ_   512
#define SIZE_ 512
#define H_    8
#define HD_   64
#define SCALE_ 0.125f   // 1/sqrt(64)

typedef __attribute__((ext_vector_type(16))) __bf16 v16bf;
typedef __attribute__((ext_vector_type(8)))  __bf16 v8bf;
typedef __attribute__((ext_vector_type(8)))  float  v8f;
typedef __attribute__((ext_vector_type(4)))  unsigned int u32x4;
typedef __attribute__((ext_vector_type(8)))  unsigned int u32x8;

// float -> bf16 (round to nearest even), as raw u16
static __device__ __forceinline__ unsigned short f2bf(float f) {
  union { float f; unsigned int u; } c; c.f = f;
  unsigned int u = c.u;
  u += 0x7fffu + ((u >> 16) & 1u);
  return (unsigned short)(u >> 16);
}

// Load a 16x32 bf16 A-style WMMA fragment from LDS (row-major, `stride` elems).
// Per CDNA5 ISA layout: lanes 0-15 hold K 0-7 & 16-23, lanes 16-31 hold K 8-15 & 24-31.
static __device__ __forceinline__ v16bf ld_frag(const unsigned short* s, int stride,
                                                int row0, int k0) {
  const int lane = threadIdx.x & 31;
  const int r  = row0 + (lane & 15);
  const int kk = k0 + ((lane & 16) ? 8 : 0);
  const unsigned short* p = s + r * stride + kk;
  v8bf lo = *(const v8bf*)(p);
  v8bf hi = *(const v8bf*)(p + 16);
  v16bf out;
#pragma unroll
  for (int i = 0; i < 8; ++i) { out[i] = lo[i]; out[i + 8] = hi[i]; }
  return out;
}

static __device__ __forceinline__ v8f wmma_bf16(v16bf a, v16bf b, v8f c) {
  // (neg_a, A, neg_b, B, c_mod, C, reuse_a, reuse_b)
  return __builtin_amdgcn_wmma_f32_16x16x32_bf16(false, a, false, b, (short)0, c,
                                                 false, false);
}

// ---------------------------------------------------------------------------
// Tensor Data Mover: 2D tile load, bf16 elements, into contiguous LDS.
// Builds the D# per CDNA5 ISA ch.8 (group0 128b + group1 256b; groups 2/3 NULL
// for 2D tensors) and issues TENSOR_LOAD_TO_LDS via inline asm (portable across
// toolchains whose clang builtins differ in arity). Tracked by TENSORcnt.
// ---------------------------------------------------------------------------
static __device__ __forceinline__ void tdm_load_2d_bf16(unsigned lds_off,
                                                        const unsigned short* gptr,
                                                        int cols, int rows,
                                                        int gstride_elems) {
  const unsigned long long ga = (unsigned long long)(size_t)gptr;
  u32x4 g0;
  g0[0] = 1u;                                              // count=1 user D#
  g0[1] = lds_off;                                         // lds_addr (bytes)
  g0[2] = (unsigned)(ga & 0xffffffffu);                    // global_addr[31:0]
  g0[3] = (unsigned)((ga >> 32) & 0x01ffffffu) | (2u << 30); // addr[56:32]|type=2
  u32x8 g1;
  g1[0] = 1u << 16;                               // data_size=1 (2B); mask/flags=0
  g1[1] = ((unsigned)cols & 0xffffu) << 16;       // tensor_dim0[15:0] (bits 63:48)
  g1[2] = (((unsigned)cols >> 16) & 0xffffu)      // tensor_dim0[31:16]
        | (((unsigned)rows & 0xffffu) << 16);     // tensor_dim1[15:0]
  g1[3] = (((unsigned)rows >> 16) & 0xffffu)      // tensor_dim1[31:16]
        | (((unsigned)cols & 0xffffu) << 16);     // tile_dim0
  g1[4] = (unsigned)rows & 0xffffu;               // tile_dim1 (tile_dim2=0 -> 2D)
  g1[5] = (unsigned)gstride_elems;                // tensor_dim0_stride[31:0]
  g1[6] = 0u;                                     // stride0 hi | stride1 lo
  g1[7] = 0u;                                     // stride1 hi
  asm volatile("tensor_load_to_lds %0, %1" :: "s"(g0), "s"(g1) : "memory");
}

// ---------------------------------------------------------------------------
// GEMM C = A[M,512] @ W^T + bias, epilogue selected at compile time:
//  MODE 0: Cf (fp32) + Cbf (bf16)           -- q projection
//  MODE 1: Cbf (bf16) only                  -- k projection
//  MODE 2: CbfT (bf16, [B,H,HD,LKV]) only   -- v projection (pre-transposed)
//  MODE 3: Cf = residual + relu(proj)       -- output projection
// ---------------------------------------------------------------------------
template <int MODE>
static __device__ __forceinline__ void epilogue_tile(v8f a, int gm0, int gn, int mbase,
                                                     const float* __restrict__ bias,
                                                     float* __restrict__ Cf,
                                                     unsigned short* __restrict__ Cbf,
                                                     unsigned short* __restrict__ CbfT,
                                                     const float* __restrict__ residual) {
  const float bb = bias[gn];
  if (MODE == 2) {
    // gm = b*512 + kv, gn = h*64 + d ; 8 consecutive kv -> one 16B store
    union { uint4 v4; unsigned short s[8]; } t;
#pragma unroll
    for (int r = 0; r < 8; ++r) t.s[r] = f2bf(a[r] + bb);
    const int bidx = gm0 >> 9;
    const size_t offT = ((size_t)(bidx * H_ + (gn >> 6)) * HD_ + (gn & 63)) * 512
                        + (size_t)((gm0 & 511) + mbase);
    *(uint4*)(CbfT + offT) = t.v4;
    return;
  }
#pragma unroll
  for (int r = 0; r < 8; ++r) {
    const size_t off = (size_t)(gm0 + mbase + r) * SIZE_ + gn;
    const float val = a[r] + bb;
    if (MODE == 3) {
      Cf[off] = residual[off] + (val > 0.f ? val : 0.f);
    } else if (MODE == 0) {
      Cf[off]  = val;
      Cbf[off] = f2bf(val);
    } else {  // MODE == 1
      Cbf[off] = f2bf(val);
    }
  }
}

template <int MODE>
__global__ __launch_bounds__(128) void gemm_xwT(const float* __restrict__ A,
                                                const float* __restrict__ W,
                                                const float* __restrict__ bias,
                                                float* __restrict__ Cf,
                                                unsigned short* __restrict__ Cbf,
                                                unsigned short* __restrict__ CbfT,
                                                const float* __restrict__ residual) {
  __shared__ __align__(16) unsigned short sA[64 * 32];
  __shared__ __align__(16) unsigned short sB[64 * 32];
  const int m0 = blockIdx.x * 64;
  const int n0 = blockIdx.y * 64;
  const int tid  = threadIdx.x;
  const int wave = tid >> 5;
  const int wm = (wave & 1) * 32;
  const int wn = (wave >> 1) * 32;

  v8f acc00 = {}, acc01 = {}, acc10 = {}, acc11 = {};

  for (int k0 = 0; k0 < SIZE_; k0 += 32) {
    // Stage 64x32 fp32->bf16 tiles of A and of W (B^T == W since C = A @ W^T)
#pragma unroll
    for (int it = 0; it < 4; ++it) {
      const int v   = tid + it * 128;   // 512 float4 vectors per tile
      const int row = v >> 3;
      const int c4  = (v & 7) * 4;
      float4 fa = *(const float4*)(A + (size_t)(m0 + row) * SIZE_ + k0 + c4);
      unsigned short* da = sA + row * 32 + c4;
      da[0] = f2bf(fa.x); da[1] = f2bf(fa.y); da[2] = f2bf(fa.z); da[3] = f2bf(fa.w);
      float4 fb = *(const float4*)(W + (size_t)(n0 + row) * SIZE_ + k0 + c4);
      unsigned short* db = sB + row * 32 + c4;
      db[0] = f2bf(fb.x); db[1] = f2bf(fb.y); db[2] = f2bf(fb.z); db[3] = f2bf(fb.w);
    }
    __syncthreads();
    v16bf a0 = ld_frag(sA, 32, wm, 0);
    v16bf a1 = ld_frag(sA, 32, wm + 16, 0);
    v16bf b0 = ld_frag(sB, 32, wn, 0);
    v16bf b1 = ld_frag(sB, 32, wn + 16, 0);
    acc00 = wmma_bf16(a0, b0, acc00);
    acc01 = wmma_bf16(a0, b1, acc01);
    acc10 = wmma_bf16(a1, b0, acc10);
    acc11 = wmma_bf16(a1, b1, acc11);
    __syncthreads();
  }

  const int lane  = tid & 31;
  const int nloc  = lane & 15;
  const int mbase = (lane & 16) ? 8 : 0;
  epilogue_tile<MODE>(acc00, m0 + wm,      n0 + wn      + nloc, mbase, bias, Cf, Cbf, CbfT, residual);
  epilogue_tile<MODE>(acc01, m0 + wm,      n0 + wn + 16 + nloc, mbase, bias, Cf, Cbf, CbfT, residual);
  epilogue_tile<MODE>(acc10, m0 + wm + 16, n0 + wn      + nloc, mbase, bias, Cf, Cbf, CbfT, residual);
  epilogue_tile<MODE>(acc11, m0 + wm + 16, n0 + wn + 16 + nloc, mbase, bias, Cf, Cbf, CbfT, residual);
}

// ---------------------------------------------------------------------------
// One block per (b, h, 16-query-row tile). 128 threads = 4 waves.
// oh = qh + (tanh(S)*softmax(sqrt(S^2+0.01))) @ V,  S = q k^T * SCALE.
// K / V / Q tiles are DMA'd into LDS by the Tensor Data Mover (wave 0 issues,
// waits TENSORcnt, workgroup barrier publishes).
// ---------------------------------------------------------------------------
__global__ __launch_bounds__(128) void attn_kernel(const float* __restrict__ qf,
                                                   const unsigned short* __restrict__ qbf,
                                                   const unsigned short* __restrict__ kbf,
                                                   const unsigned short* __restrict__ vbfT,
                                                   float* __restrict__ out) {
  __shared__ __align__(16) unsigned short sQ[16 * 64];    //  2 KB
  __shared__ __align__(16) unsigned short sKV[64 * 64];   //  8 KB (K chunk / V^T chunk)
  __shared__ __align__(16) float sS[16 * 512];            // 32 KB (scores -> P fp32)
  __shared__ __align__(16) unsigned short sP[16 * 512];   // 16 KB (normalized P, bf16)
  __shared__ float sPart[16 * 8];
  __shared__ float sInv[16];

  const int tid  = threadIdx.x;
  const int lane = tid & 31;
  const int w    = tid >> 5;
  const int blk  = blockIdx.x;
  const int b    = blk >> 8;          // H_*32 = 256 blocks per batch
  const int h    = (blk >> 5) & 7;
  const int q0   = (blk & 31) * 16;
  const size_t rowBase = (size_t)b * LQ_ * SIZE_;
  const int hoff = h * HD_;

  const unsigned ldsQ  = (unsigned)(size_t)(void*)sQ;
  const unsigned ldsKV = (unsigned)(size_t)(void*)sKV;

  // DMA the 16x64 bf16 Q tile
  if (w == 0) {
    tdm_load_2d_bf16(ldsQ, qbf + rowBase + (size_t)q0 * SIZE_ + hoff, 64, 16, SIZE_);
    __builtin_amdgcn_s_wait_tensorcnt(0);
  }
  __syncthreads();
  const v16bf qa0 = ld_frag(sQ, 64, 0, 0);
  const v16bf qa1 = ld_frag(sQ, 64, 0, 32);

  // ---- S = Q K^T over 8 chunks of 64 kv rows; wave w owns 16 cols per chunk ----
  for (int kc = 0; kc < 8; ++kc) {
    if (w == 0) {
      tdm_load_2d_bf16(ldsKV, kbf + rowBase + (size_t)(kc * 64) * SIZE_ + hoff,
                       64, 64, SIZE_);
      __builtin_amdgcn_s_wait_tensorcnt(0);
    }
    __syncthreads();
    v16bf b0 = ld_frag(sKV, 64, w * 16, 0);
    v16bf b1 = ld_frag(sKV, 64, w * 16, 32);
    v8f sacc = {};
    sacc = wmma_bf16(qa0, b0, sacc);
    sacc = wmma_bf16(qa1, b1, sacc);
    const int nloc = lane & 15, mbase = (lane & 16) ? 8 : 0;
#pragma unroll
    for (int r = 0; r < 8; ++r)
      sS[(mbase + r) * 512 + kc * 64 + w * 16 + nloc] = sacc[r];
    __syncthreads();
  }

  // ---- signed softmax: P = tanh(s)*exp(sqrt(s^2+.01)), rowsum over exp only ----
  {
    const int r = tid >> 3, g = tid & 7;
    float se = 0.f;
    for (int j = g * 64; j < g * 64 + 64; ++j) {
      const float s = sS[r * 512 + j] * SCALE_;
      const float e = __expf(sqrtf(s * s + 0.01f));
      sS[r * 512 + j] = tanhf(s) * e;
      se += e;
    }
    sPart[r * 8 + g] = se;
  }
  __syncthreads();
  if (tid < 16) {
    float t = 0.f;
#pragma unroll
    for (int g = 0; g < 8; ++g) t += sPart[tid * 8 + g];
    sInv[tid] = 1.f / t;
  }
  __syncthreads();
  {
    const int r = tid >> 3, g = tid & 7;
    const float inv = sInv[r];
    for (int j = g * 64; j < g * 64 + 64; ++j)
      sP[r * 512 + j] = f2bf(sS[r * 512 + j] * inv);
  }
  __syncthreads();

  // ---- O = P @ V; wave w owns head-dim cols [w*16, w*16+16) ----
  // V is stored pre-transposed: vbfT[((b*H + h)*HD + d) * 512 + kv]
  const unsigned short* vbase = vbfT + ((size_t)(b * H_ + h) * HD_) * 512;
  v8f oacc = {};
  for (int kc = 0; kc < 8; ++kc) {
    if (w == 0) {
      tdm_load_2d_bf16(ldsKV, vbase + kc * 64, 64, 64, 512);
      __builtin_amdgcn_s_wait_tensorcnt(0);
    }
    __syncthreads();
    v16bf pa0 = ld_frag(sP, 512, 0, kc * 64);
    v16bf pa1 = ld_frag(sP, 512, 0, kc * 64 + 32);
    v16bf vb0 = ld_frag(sKV, 64, w * 16, 0);
    v16bf vb1 = ld_frag(sKV, 64, w * 16, 32);
    oacc = wmma_bf16(pa0, vb0, oacc);
    oacc = wmma_bf16(pa1, vb1, oacc);
    __syncthreads();
  }

  // ---- residual + store ----
  {
    const int nloc = lane & 15, mbase = (lane & 16) ? 8 : 0;
    const int d = w * 16 + nloc;
#pragma unroll
    for (int r = 0; r < 8; ++r) {
      const size_t off = rowBase + (size_t)(q0 + mbase + r) * SIZE_ + hoff + d;
      out[off] = qf[off] + oacc[r];
    }
  }
}

// LayerNorm over last dim (512). One wave per row, 8 rows per 256-thread block.
__global__ __launch_bounds__(256) void ln_kernel(const float* __restrict__ x,
                                                 const float* __restrict__ gw,
                                                 const float* __restrict__ gb,
                                                 float* __restrict__ y) {
  const int row  = blockIdx.x * 8 + (threadIdx.x >> 5);
  const int lane = threadIdx.x & 31;
  const float* xr = x + (size_t)row * SIZE_;
  float4 v[4];
  float sum = 0.f, sq = 0.f;
#pragma unroll
  for (int i = 0; i < 4; ++i) {
    v[i] = *(const float4*)(xr + i * 128 + lane * 4);
    sum += v[i].x + v[i].y + v[i].z + v[i].w;
    sq  += v[i].x * v[i].x + v[i].y * v[i].y + v[i].z * v[i].z + v[i].w * v[i].w;
  }
#pragma unroll
  for (int o = 16; o > 0; o >>= 1) {
    sum += __shfl_xor(sum, o, 32);
    sq  += __shfl_xor(sq, o, 32);
  }
  const float mu  = sum * (1.f / 512.f);
  const float var = sq * (1.f / 512.f) - mu * mu;
  const float rs  = rsqrtf(var + 1e-5f);
  float* yr = y + (size_t)row * SIZE_;
#pragma unroll
  for (int i = 0; i < 4; ++i) {
    const int c = i * 128 + lane * 4;
    float4 o4;
    o4.x = (v[i].x - mu) * rs * gw[c + 0] + gb[c + 0];
    o4.y = (v[i].y - mu) * rs * gw[c + 1] + gb[c + 1];
    o4.z = (v[i].z - mu) * rs * gw[c + 2] + gb[c + 2];
    o4.w = (v[i].w - mu) * rs * gw[c + 3] + gb[c + 3];
    *(float4*)(yr + c) = o4;
  }
}

extern "C" void kernel_launch(void* const* d_in, const int* in_sizes, int n_in,
                              void* d_out, int out_size, void* d_ws, size_t ws_size,
                              hipStream_t stream) {
  (void)in_sizes; (void)n_in; (void)out_size; (void)ws_size;
  const float* query     = (const float*)d_in[0];
  const float* key_value = (const float*)d_in[1];
  const float* Wq = (const float*)d_in[2];
  const float* bq = (const float*)d_in[3];
  const float* Wk = (const float*)d_in[4];
  const float* bk = (const float*)d_in[5];
  const float* Wv = (const float*)d_in[6];
  const float* bv = (const float*)d_in[7];
  const float* Wo = (const float*)d_in[8];
  const float* bo = (const float*)d_in[9];
  const float* ln0w = (const float*)d_in[10];
  const float* ln0b = (const float*)d_in[11];
  const float* ln1w = (const float*)d_in[12];
  const float* ln1b = (const float*)d_in[13];

  const size_t NF = (size_t)B_ * LQ_ * SIZE_;  // 8,388,608 elements
  char* ws = (char*)d_ws;
  float*          qf   = (float*)(ws);                     // fp32 q (residual)   32 MB
  unsigned short* qbf  = (unsigned short*)(ws + NF * 4);   // bf16 q              16 MB
  unsigned short* kbf  = (unsigned short*)(ws + NF * 6);   // bf16 k              16 MB
  unsigned short* vbfT = (unsigned short*)(ws + NF * 8);   // bf16 v^T [B,H,HD,L] 16 MB
  float*          oh   = (float*)(ws + NF * 10);           // attn out            32 MB
  float*          ln0f = (float*)(ws + NF * 14);           // LN0 out             32 MB
  // total workspace: NF*18 bytes = 144 MB; qf region reused for projection-2 out

  const dim3 gg(256, 8);  // (16384/64) x (512/64)
  gemm_xwT<0><<<gg, 128, 0, stream>>>(query, Wq, bq, qf, qbf, nullptr, nullptr);
  gemm_xwT<1><<<gg, 128, 0, stream>>>(key_value, Wk, bk, nullptr, kbf, nullptr, nullptr);
  gemm_xwT<2><<<gg, 128, 0, stream>>>(key_value, Wv, bv, nullptr, nullptr, vbfT, nullptr);
  attn_kernel<<<B_ * H_ * (LQ_ / 16), 128, 0, stream>>>(qf, qbf, kbf, vbfT, oh);
  ln_kernel<<<(B_ * LQ_) / 8, 256, 0, stream>>>(oh, ln0w, ln0b, ln0f);
  gemm_xwT<3><<<gg, 128, 0, stream>>>(ln0f, Wo, bo, qf /*reuse as out2*/, nullptr,
                                      nullptr, ln0f);
  ln_kernel<<<(B_ * LQ_) / 8, 256, 0, stream>>>(qf, ln1w, ln1b, (float*)d_out);
}
